// SURFMNetLoss_90915867722500
// MI455X (gfx1250) — matrix operand order repaired and checked by reading, not compiled, not run on hardware
//
#include <hip/hip_runtime.h>
#include <hip/hip_bf16.h>

typedef __attribute__((ext_vector_type(16))) _Float16 v16h;
typedef __attribute__((ext_vector_type(8)))  _Float16 v8h;
typedef __attribute__((ext_vector_type(8)))  float    v8f;
typedef __attribute__((ext_vector_type(4)))  float    v4f;
typedef __attribute__((ext_vector_type(2)))  float    v2f;

#define N_   8
#define V_   5000
#define C_   128
#define K_   128
#define D_   25
#define V_P  5024   // V padded to multiple of 32 (zero-filled)

// ---------------------------------------------------------------------------
// Prep 1: evecs_trans (N,K,V) f32 -> f16, padded rows. 2048 rows total (2 sides).
// ---------------------------------------------------------------------------
__global__ __launch_bounds__(256) void prep_copy(
    const float* __restrict__ et1, const float* __restrict__ et2,
    _Float16* __restrict__ EtH)
{
  const int r = blockIdx.x;                 // 0..2047
  const int side = r >> 10;
  const int rem = r & 1023;                 // n*K + k
  const float* src = (side ? et2 : et1) + (size_t)rem * V_;
  _Float16* dst = EtH + (size_t)r * V_P;
  for (int g = threadIdx.x; g < V_P / 8; g += 256) {
    const int v0 = g * 8;                   // V_ % 8 == 0 -> groups all-or-nothing
    v8h h = {};
    if (v0 < V_) {
      v4f x = *(const v4f*)(src + v0);
      v4f y = *(const v4f*)(src + v0 + 4);
#pragma unroll
      for (int q = 0; q < 4; ++q) { h[q] = (_Float16)x[q]; h[q + 4] = (_Float16)y[q]; }
    }
    *(v8h*)(dst + v0) = h;
  }
}

// ---------------------------------------------------------------------------
// Prep 2: evecs (N,V,K) f32 -> f16 transposed [side,n][j][v], padded.
// 32x32 LDS tile transpose, coalesced read and write.
// ---------------------------------------------------------------------------
__global__ __launch_bounds__(256) void prep_tr(
    const float* __restrict__ e1, const float* __restrict__ e2,
    _Float16* __restrict__ ETH)
{
  const int vt = blockIdx.x * 32, jt = blockIdx.y * 32;
  const int z = blockIdx.z;                 // side*8+n
  const int side = z >> 3, n = z & 7;
  const float* src = (side ? e2 : e1) + (size_t)n * V_ * K_;
  _Float16* dst = ETH + (size_t)z * K_ * V_P;
  __shared__ float tile[32][33];
  const int c = threadIdx.x & 31, r0 = threadIdx.x >> 5;
#pragma unroll
  for (int i = 0; i < 4; ++i) {
    int r = r0 + i * 8;
    int v = vt + r;
    tile[r][c] = (v < V_) ? src[(size_t)v * K_ + (jt + c)] : 0.f;
  }
  __syncthreads();
#pragma unroll
  for (int i = 0; i < 4; ++i) {
    int r = r0 + i * 8;
    dst[(size_t)(jt + r) * V_P + (vt + c)] = (_Float16)tile[c][r];
  }
}

// ---------------------------------------------------------------------------
// Prep 3: FG[side*200 + n*D + d][v] = f16(feat[n,v,desc[d]]), padded.
// ---------------------------------------------------------------------------
__global__ __launch_bounds__(256) void prep_fg(
    const float* __restrict__ f1, const float* __restrict__ f2,
    const int* __restrict__ desc, _Float16* __restrict__ FG)
{
  const int v = blockIdx.x * 256 + threadIdx.x;
  const int z = blockIdx.y;                 // side*8+n
  const int side = z >> 3, n = z & 7;
  if (v >= V_P) return;
  const float* src = (side ? f2 : f1) + (size_t)n * V_ * C_;
#pragma unroll 1
  for (int d = 0; d < D_; ++d) {
    float x = (v < V_) ? src[(size_t)v * C_ + desc[d]] : 0.f;
    FG[((size_t)side * 200 + n * D_ + d) * V_P + v] = (_Float16)x;
  }
}

// ---------------------------------------------------------------------------
// Kernel 1: M[nd] = (Et .* f) @ E (prepped f16 layouts).
// Wave owns 32 k-rows x 64 j-cols (2x4 accumulators): 8 WMMAs per chunk for
// only 14 loads (A frags reused across 4 j-tiles, B frags across 2 k-tiles).
// sched_group_barrier pins: [14 vmem] -> [16 pk_mul] -> [8 wmma] so the WMMA
// drain overlaps outstanding loads with partial loadcnt waits.
// ---------------------------------------------------------------------------
__global__ __launch_bounds__(256) void gemmM_kernel(
    const _Float16* __restrict__ EtH, const _Float16* __restrict__ ETH,
    const _Float16* __restrict__ FG,
    float* __restrict__ M1, float* __restrict__ M2)
{
  const int nd = blockIdx.x;                // 0..199
  const int side = blockIdx.y;
  const int n = nd / D_;
  const _Float16* A = EtH + ((size_t)side * N_ + n) * K_ * V_P;  // [k][v]
  const _Float16* B = ETH + ((size_t)side * N_ + n) * K_ * V_P;  // [j][v]
  const _Float16* F = FG + ((size_t)side * 200 + nd) * V_P;      // [v]
  float* Mo = (side ? M2 : M1) + (size_t)nd * (K_ * K_);

  const int tid = threadIdx.x;
  const int lane = tid & 31, wave = tid >> 5;
  const int lm = lane & 15, hf = lane >> 4;
  const int kq = wave & 3;                  // k quarter: rows [kq*32, +32)
  const int jh = wave >> 2;                 // j half:    cols [jh*64, +64)

  const _Float16* Arow0 = A + (size_t)(kq * 32 + lm) * V_P + hf * 8;
  const _Float16* Arow1 = Arow0 + (size_t)16 * V_P;
  const _Float16* Frow  = F + hf * 8;
  const _Float16* Bbase = B + (size_t)(jh * 64 + lm) * V_P + hf * 8;

  v8f acc[2][4] = {};

#pragma unroll 1
  for (int v0 = 0; v0 < V_P; v0 += 32) {
    // ---- 14 loads: 2 A fragments (4), F (2), 4 B fragments (8) ----
    v8h ea0 = *(const v8h*)(Arow0 + v0);
    v8h ea1 = *(const v8h*)(Arow0 + v0 + 16);
    v8h eb0 = *(const v8h*)(Arow1 + v0);
    v8h eb1 = *(const v8h*)(Arow1 + v0 + 16);
    v8h f0  = *(const v8h*)(Frow + v0);
    v8h f1  = *(const v8h*)(Frow + v0 + 16);
    v8h b0[4], b1[4];
#pragma unroll
    for (int jt = 0; jt < 4; ++jt) {
      const _Float16* Brow = Bbase + (size_t)jt * 16 * V_P + v0;
      b0[jt] = *(const v8h*)Brow;
      b1[jt] = *(const v8h*)(Brow + 16);
    }
    // ---- diag(f) fold into both A fragments (16 v_pk_mul_f16) ----
    v16h a0, a1;
#pragma unroll
    for (int q = 0; q < 8; ++q) {
      a0[q] = ea0[q] * f0[q]; a0[q + 8] = ea1[q] * f1[q];
      a1[q] = eb0[q] * f0[q]; a1[q + 8] = eb1[q] * f1[q];
    }
    // ---- 8 WMMAs (2 k-tiles x 4 j-tiles) ----
#pragma unroll
    for (int jt = 0; jt < 4; ++jt) {
      v16h b;
#pragma unroll
      for (int q = 0; q < 8; ++q) { b[q] = b0[jt][q]; b[q + 8] = b1[jt][q]; }
      acc[0][jt] = __builtin_amdgcn_wmma_f32_16x16x32_f16(
          false, a0, false, b, (short)0, acc[0][jt], false, false);
      acc[1][jt] = __builtin_amdgcn_wmma_f32_16x16x32_f16(
          false, a1, false, b, (short)0, acc[1][jt], false, false);
    }
#if __has_builtin(__builtin_amdgcn_sched_group_barrier)
    __builtin_amdgcn_sched_group_barrier(0x020, 14, 0); // VMEM reads first
    __builtin_amdgcn_sched_group_barrier(0x002, 16, 0); // then pk_mul fold
    __builtin_amdgcn_sched_group_barrier(0x008, 8, 0);  // then WMMA drain
#endif
  }

  // C/D layout: lane<16 -> col=lane, rows r; lane>=16 -> col=lane-16, rows 8+r
#pragma unroll
  for (int t = 0; t < 2; ++t)
#pragma unroll
    for (int jt = 0; jt < 4; ++jt)
#pragma unroll
      for (int r = 0; r < 8; ++r)
        Mo[(kq * 32 + t * 16 + hf * 8 + r) * K_ + jh * 64 + jt * 16 + lm] =
            acc[t][jt][r];
}

// ---------------------------------------------------------------------------
// Kernel 2: l_pre partials: sum((C@Ma - Mb@C)^2) per (n,d,side), f32 WMMA.
// ---------------------------------------------------------------------------
__global__ __launch_bounds__(256) void pre_kernel(
    const float* __restrict__ C12, const float* __restrict__ C21,
    const float* __restrict__ M1,  const float* __restrict__ M2,
    float* __restrict__ parts)
{
  const int nd = blockIdx.x, y = blockIdx.y;
  const float* Cm = y ? C21 : C12;
  const float* Ma = y ? M2  : M1;   // s = C @ Ma
  const float* Mb = y ? M1  : M2;   // t = Mb @ C
  const int n = nd / D_;
  Cm += (size_t)n * K_ * K_;
  Ma += (size_t)nd * K_ * K_;
  Mb += (size_t)nd * K_ * K_;

  const int tid = threadIdx.x, lane = tid & 31, wave = tid >> 5;
  const int lm = lane & 15, hf = lane >> 4;
  float part = 0.f;

#pragma unroll 1
  for (int t = wave; t < 64; t += 8) {
    const int it = (t >> 3) * 16, ct = (t & 7) * 16;
    v8f aS = {}, aT = {};
#pragma unroll 4
    for (int j0 = 0; j0 < K_; j0 += 4) {
      const int jb = j0 + 2 * hf;
      v2f a1 = *(const v2f*)&Cm[(it + lm) * K_ + jb];
      v2f b1; b1[0] = Ma[jb * K_ + ct + lm]; b1[1] = Ma[(jb + 1) * K_ + ct + lm];
      aS = __builtin_amdgcn_wmma_f32_16x16x4_f32(false, a1, false, b1, (short)0, aS, false, false);
      v2f a2 = *(const v2f*)&Mb[(it + lm) * K_ + jb];
      v2f b2; b2[0] = Cm[jb * K_ + ct + lm]; b2[1] = Cm[(jb + 1) * K_ + ct + lm];
      aT = __builtin_amdgcn_wmma_f32_16x16x4_f32(false, a2, false, b2, (short)0, aT, false, false);
    }
#pragma unroll
    for (int r = 0; r < 8; ++r) { float dd = aS[r] - aT[r]; part += dd * dd; }
  }

  __shared__ float red[256];
  red[tid] = part; __syncthreads();
  for (int s = 128; s > 0; s >>= 1) { if (tid < s) red[tid] += red[tid + s]; __syncthreads(); }
  if (tid == 0) parts[y * 200 + nd] = red[0];
}

// ---------------------------------------------------------------------------
// Kernel 3: ||P - I||^2 for y=0:C12@C21 y=1:C21@C12 y=2:C12^T@C12 y=3:C21^T@C21
// ---------------------------------------------------------------------------
__global__ __launch_bounds__(256) void bo_kernel(
    const float* __restrict__ C12, const float* __restrict__ C21,
    float* __restrict__ parts)
{
  const int n = blockIdx.x, y = blockIdx.y;
  const float* Araw = (y == 1 || y == 3) ? C21 : C12;
  const float* Braw = (y == 0) ? C21 : (y == 1) ? C12 : Araw;
  const bool trans = (y >= 2);
  const float* A = Araw + (size_t)n * K_ * K_;
  const float* B = Braw + (size_t)n * K_ * K_;

  const int tid = threadIdx.x, lane = tid & 31, wave = tid >> 5;
  const int lm = lane & 15, hf = lane >> 4;
  float part = 0.f;

#pragma unroll 1
  for (int t = wave; t < 64; t += 8) {
    const int it = (t >> 3) * 16, ct = (t & 7) * 16;
    v8f accP = {};
#pragma unroll 4
    for (int j0 = 0; j0 < K_; j0 += 4) {
      const int jb = j0 + 2 * hf;
      v2f a;
      if (trans) { a[0] = A[jb * K_ + it + lm]; a[1] = A[(jb + 1) * K_ + it + lm]; }
      else       { a = *(const v2f*)&A[(it + lm) * K_ + jb]; }
      v2f b; b[0] = B[jb * K_ + ct + lm]; b[1] = B[(jb + 1) * K_ + ct + lm];
      accP = __builtin_amdgcn_wmma_f32_16x16x4_f32(false, a, false, b, (short)0, accP, false, false);
    }
#pragma unroll
    for (int r = 0; r < 8; ++r) {
      int row = it + hf * 8 + r, col = ct + lm;
      float p = accP[r] - (row == col ? 1.f : 0.f);
      part += p * p;
    }
  }

  __shared__ float red[256];
  red[tid] = part; __syncthreads();
  for (int s = 128; s > 0; s >>= 1) { if (tid < s) red[tid] += red[tid + s]; __syncthreads(); }
  if (tid == 0) parts[y * N_ + n] = red[0];
}

// ---------------------------------------------------------------------------
// Kernel 4: Laplacian term: sum (C[i,j]*(ea[j]-eb[i]))^2
// ---------------------------------------------------------------------------
__global__ __launch_bounds__(256) void lap_kernel(
    const float* __restrict__ C12, const float* __restrict__ C21,
    const float* __restrict__ ev1, const float* __restrict__ ev2,
    float* __restrict__ parts)
{
  const int y = blockIdx.y, b = blockIdx.x, tid = threadIdx.x;
  const float* Cm = y ? C21 : C12;
  const float* ea = y ? ev2 : ev1;   // column factor
  const float* eb = y ? ev1 : ev2;   // row factor
  float part = 0.f;
#pragma unroll
  for (int i = 0; i < 4; ++i) {
    int idx = b * 1024 + i * 256 + tid;   // < 8*128*128
    int n = idx >> 14, r = (idx >> 7) & 127, c = idx & 127;
    float cv = Cm[idx];
    float dd = cv * (ea[n * K_ + c] - eb[n * K_ + r]);
    part += dd * dd;
  }
  __shared__ float red[256];
  red[tid] = part; __syncthreads();
  for (int s = 128; s > 0; s >>= 1) { if (tid < s) red[tid] += red[tid + s]; __syncthreads(); }
  if (tid == 0) parts[y * 128 + b] = red[0];
}

// ---------------------------------------------------------------------------
// Kernel 5: deterministic final reduction + weights
// ---------------------------------------------------------------------------
__global__ __launch_bounds__(256) void fin_kernel(
    const float* __restrict__ partsPre, const float* __restrict__ partsBO,
    const float* __restrict__ partsLap, float* __restrict__ out)
{
  __shared__ float red[256];
  const int tid = threadIdx.x;

  float s = 0.f;
  for (int i = tid; i < 400; i += 256) s += partsPre[i];
  red[tid] = s; __syncthreads();
  for (int st = 128; st > 0; st >>= 1) { if (tid < st) red[tid] += red[tid + st]; __syncthreads(); }
  float Spre = red[0]; __syncthreads();

  red[tid] = (tid < 16) ? partsBO[tid] : 0.f; __syncthreads();
  for (int st = 128; st > 0; st >>= 1) { if (tid < st) red[tid] += red[tid + st]; __syncthreads(); }
  float Sbij = red[0]; __syncthreads();

  red[tid] = (tid < 16) ? partsBO[16 + tid] : 0.f; __syncthreads();
  for (int st = 128; st > 0; st >>= 1) { if (tid < st) red[tid] += red[tid + st]; __syncthreads(); }
  float Sorth = red[0]; __syncthreads();

  red[tid] = partsLap[tid]; __syncthreads();
  for (int st = 128; st > 0; st >>= 1) { if (tid < st) red[tid] += red[tid + st]; __syncthreads(); }
  float Slap = red[0];

  if (tid == 0) {
    out[0] = 1000.f   * (Sbij  / (float)N_);
    out[1] = 1000.f   * (Sorth / (float)N_);
    out[2] =            (Slap  / (float)N_);
    out[3] = 100000.f * (Spre  / (float)(N_ * K_ * K_));
  }
}

extern "C" void kernel_launch(void* const* d_in, const int* in_sizes, int n_in,
                              void* d_out, int out_size, void* d_ws, size_t ws_size,
                              hipStream_t stream) {
  const float* C12  = (const float*)d_in[0];
  const float* C21  = (const float*)d_in[1];
  const float* f1   = (const float*)d_in[2];
  const float* f2   = (const float*)d_in[3];
  const float* e1   = (const float*)d_in[4];
  const float* e2   = (const float*)d_in[5];
  const float* et1  = (const float*)d_in[6];
  const float* et2  = (const float*)d_in[7];
  const float* ev1  = (const float*)d_in[8];
  const float* ev2  = (const float*)d_in[9];
  const int*   desc = (const int*)d_in[10];

  // workspace layout: [partials f32][M1 f32][M2 f32][EtH f16][ETH f16][FG f16]
  float* pPre = (float*)d_ws;                     // 400
  float* pBO  = pPre + 400;                       // 32
  float* pLap = pBO + 32;                         // 256 (pad to 688 -> 16B mult)
  float* M1   = pPre + 688;                       // 200*16384
  float* M2   = M1 + (size_t)200 * 16384;
  _Float16* EtH = (_Float16*)(M2 + (size_t)200 * 16384);   // 2048 * V_P
  _Float16* ETH = EtH + (size_t)2048 * V_P;                // 2048 * V_P
  _Float16* FG  = ETH + (size_t)2048 * V_P;                // 400  * V_P

  prep_copy  <<<dim3(2048),       dim3(256), 0, stream>>>(et1, et2, EtH);
  prep_tr    <<<dim3(157, 4, 16), dim3(256), 0, stream>>>(e1, e2, ETH);
  prep_fg    <<<dim3(20, 16),     dim3(256), 0, stream>>>(f1, f2, desc, FG);
  gemmM_kernel<<<dim3(200, 2),    dim3(256), 0, stream>>>(EtH, ETH, FG, M1, M2);
  pre_kernel <<<dim3(200, 2),     dim3(256), 0, stream>>>(C12, C21, M1, M2, pPre);
  bo_kernel  <<<dim3(N_, 4),      dim3(256), 0, stream>>>(C12, C21, pBO);
  lap_kernel <<<dim3(128, 2),     dim3(256), 0, stream>>>(C12, C21, ev1, ev2, pLap);
  fin_kernel <<<dim3(1),          dim3(256), 0, stream>>>(pPre, pBO, pLap, (float*)d_out);
}
